// Transformer_33200097198633
// MI455X (gfx1250) — matrix operand order repaired and checked by reading, not compiled, not run on hardware
//
#include <hip/hip_runtime.h>
#include <hip/hip_bf16.h>

// ---- problem constants (from reference) ----
#define H_     512
#define NH_    8
#define HD_    64
#define LE_    6
#define LD_    6
#define V_     32000
#define BB_    8
#define S_     512
#define MAXDEC_ 128
#define FF_    2048
#define BS_    (BB_*S_)
#define SCALE_ATT 0.125f   // 1/sqrt(64)

// ---- WMMA types ----
typedef __attribute__((ext_vector_type(16))) __bf16 v16bf;
typedef __attribute__((ext_vector_type(8)))  float  v8f;

union FragBF { v16bf v; unsigned int u[8]; };

__device__ __forceinline__ unsigned short f2bf(float f) {
  unsigned int u = __builtin_bit_cast(unsigned int, f);
  u += 0x7FFFu + ((u >> 16) & 1u);      // round-to-nearest-even
  return (unsigned short)(u >> 16);
}

__device__ __forceinline__ unsigned long long pack4bf(float a, float b, float c, float d) {
  unsigned int lo = (unsigned int)f2bf(a) | ((unsigned int)f2bf(b) << 16);
  unsigned int hi = (unsigned int)f2bf(c) | ((unsigned int)f2bf(d) << 16);
  return (unsigned long long)lo | ((unsigned long long)hi << 32);
}

// =====================================================================
// Tiled GEMM: C[M,N] = act(A[M,K] * B[K,N] + bias), fp32 in/out,
// bf16 WMMA inner product.
// 256 threads = 8 waves; block tile 128x128, BK=64 (2 WMMA k-steps per
// LDS fill); each wave computes 64x32 = 4x2 fragments -> 16 WMMAs per
// barrier pair. Loads are branch-free (clamped address + cndmask) and
// vectorized (b128 global, b64 LDS stores). Requires N % 128 == 0 and
// K % 64 == 0 (true for all GEMMs in this model).
// =====================================================================
#define BM_ 128
#define BN_ 128
#define BK_ 64
#define LDSK_ (BK_ + 4)   // 68 halfwords -> 136B row stride (8B aligned)

__global__ __launch_bounds__(256)
void gemm_bf16_wmma_kernel(const float* __restrict__ A, int lda,
                           const float* __restrict__ Bw, int ldb,
                           const float* __restrict__ bias,
                           float* __restrict__ C, long long ldc,
                           int M, int N, int K, int relu)
{
  __shared__ unsigned short As[BM_][LDSK_];
  __shared__ unsigned short Bs[BN_][LDSK_];   // stored transposed: Bs[n][k]

  const int tid  = threadIdx.x;
  const int lane = tid & 31;
  const int wave = tid >> 5;
  const int row0 = blockIdx.y * BM_;
  const int col0 = blockIdx.x * BN_;
  const int wm   = (wave >> 2) * 64;    // 2 wave-rows
  const int wn   = (wave & 3) * 32;     // 4 wave-cols
  const int l    = lane & 15;
  const int hh   = lane >> 4;

  // B-tile loader coords (fixed per thread)
  const int bn = tid & 127;             // n within tile
  const int bk0 = (tid >> 7) * 32;      // k half: 0 or 32

  v8f acc[4][2] = {};

  for (int kt = 0; kt < K; kt += BK_) {
    // ---- A tile: 128 rows x 64 cols, float4 loads, branch-free ----
    #pragma unroll
    for (int it = 0; it < 8; ++it) {
      int e4 = tid + it * 256;          // float4 slot: 128 rows * 16 f4/row
      int r  = e4 >> 4;
      int c4 = e4 & 15;
      int gm = row0 + r;
      int gmc = (gm < M) ? gm : (M - 1);
      float4 f = *(const float4*)(A + (size_t)gmc * lda + kt + c4 * 4);
      if (gm >= M) { f.x = 0.f; f.y = 0.f; f.z = 0.f; f.w = 0.f; }
      *(unsigned long long*)&As[r][c4 * 4] = pack4bf(f.x, f.y, f.z, f.w);
    }
    // ---- B tile: 64 k x 128 n, coalesced column loads, transposed store ----
    #pragma unroll
    for (int it = 0; it < 8; ++it) {
      int k0 = bk0 + it * 4;
      const float* bp = Bw + (size_t)(kt + k0) * ldb + (col0 + bn);
      float x0 = bp[0];
      float x1 = bp[(size_t)ldb];
      float x2 = bp[(size_t)ldb * 2];
      float x3 = bp[(size_t)ldb * 3];
      *(unsigned long long*)&Bs[bn][k0] = pack4bf(x0, x1, x2, x3);
    }
    __syncthreads();

    // ---- two WMMA k-steps over the 64-wide LDS tile ----
    #pragma unroll
    for (int ks = 0; ks < BK_; ks += 32) {
      FragBF fa[4], fb[2];
      #pragma unroll
      for (int i = 0; i < 4; ++i) {
        int m = wm + i * 16 + l;
        #pragma unroll
        for (int vv = 0; vv < 8; ++vv) {
          int k0 = ks + ((vv >> 2) << 4) + (hh << 3) + ((vv & 3) << 1);   // A layout
          fa[i].u[vv] = *(const unsigned int*)&As[m][k0];
        }
      }
      #pragma unroll
      for (int j = 0; j < 2; ++j) {
        int n = wn + j * 16 + l;
        #pragma unroll
        for (int vv = 0; vv < 8; ++vv) {
          int k0 = ks + ((vv >> 2) << 4) + (hh << 3) + ((vv & 3) << 1);   // B layout
          fb[j].u[vv] = *(const unsigned int*)&Bs[n][k0];
        }
      }
      #pragma unroll
      for (int i = 0; i < 4; ++i)
        #pragma unroll
        for (int j = 0; j < 2; ++j)
          acc[i][j] = __builtin_amdgcn_wmma_f32_16x16x32_bf16(
              false, fa[i].v, false, fb[j].v, (short)0, acc[i][j], false, false);
    }
    __syncthreads();
  }

  // epilogue: C/D layout -> element (m,n): vgpr r, lane = (m>>3)*16 + n
  #pragma unroll
  for (int i = 0; i < 4; ++i)
    #pragma unroll
    for (int j = 0; j < 2; ++j)
      #pragma unroll
      for (int r = 0; r < 8; ++r) {
        int m = row0 + wm + i * 16 + hh * 8 + r;
        int n = col0 + wn + j * 16 + l;
        if (m < M) {
          float val = acc[i][j][r];
          if (bias) val += bias[n];
          if (relu) val = fmaxf(val, 0.0f);
          C[(long long)m * ldc + n] = val;
        }
      }
}

// =====================================================================
// Embedding gather: out[t, :] = emb[tok[t], :]
// =====================================================================
__global__ __launch_bounds__(128)
void embed_kernel(const int* __restrict__ tok, const float* __restrict__ emb,
                  float* __restrict__ o)
{
  int t = blockIdx.x;
  int id = tok[t];
  #pragma unroll
  for (int i = 0; i < 4; ++i) {
    int c = threadIdx.x + i * 128;
    o[(size_t)t * H_ + c] = emb[(size_t)id * H_ + c];
  }
}

// =====================================================================
// out = LayerNorm(a + r) * g + be   (row length H_=512, 128 threads)
// safe for out == a (all reads precede writes via barriers)
// =====================================================================
__global__ __launch_bounds__(128)
void add_ln_kernel(const float* __restrict__ a, const float* __restrict__ r,
                   const float* __restrict__ g, const float* __restrict__ be,
                   float* __restrict__ o)
{
  __shared__ float red[128];
  int row = blockIdx.x, tid = threadIdx.x;
  size_t base = (size_t)row * H_;
  float v[4]; float s = 0.0f;
  #pragma unroll
  for (int i = 0; i < 4; ++i) {
    int idx = tid + i * 128;
    v[i] = a[base + idx] + r[base + idx];
    s += v[i];
  }
  red[tid] = s; __syncthreads();
  for (int st = 64; st > 0; st >>= 1) { if (tid < st) red[tid] += red[tid + st]; __syncthreads(); }
  float mean = red[0] / (float)H_;
  __syncthreads();
  float s2 = 0.0f;
  #pragma unroll
  for (int i = 0; i < 4; ++i) { float d = v[i] - mean; s2 += d * d; }
  red[tid] = s2; __syncthreads();
  for (int st = 64; st > 0; st >>= 1) { if (tid < st) red[tid] += red[tid + st]; __syncthreads(); }
  float inv = rsqrtf(red[0] / (float)H_ + 1e-5f);
  #pragma unroll
  for (int i = 0; i < 4; ++i) {
    int idx = tid + i * 128;
    o[base + idx] = (v[i] - mean) * inv * g[idx] + be[idx];
  }
}

// =====================================================================
// Encoder self-attention, fused softmax. One block = one (b,h,query row).
// Q,K,V laid out [B,S,H] with head h at columns h*64..h*64+63.
// =====================================================================
__global__ __launch_bounds__(256)
void attn_enc_kernel(const float* __restrict__ Q, const float* __restrict__ K,
                     const float* __restrict__ Vv, const int* __restrict__ src,
                     float* __restrict__ O)
{
  const int b = blockIdx.x / NH_;
  const int h = blockIdx.x % NH_;
  const int qi = blockIdx.y;
  const int tid = threadIdx.x;
  __shared__ float qv[HD_];
  __shared__ float sc[S_];
  __shared__ float red[256];
  __shared__ float part[4][HD_];

  if (tid < HD_) qv[tid] = Q[((size_t)(b * S_ + qi)) * H_ + h * HD_ + tid];
  __syncthreads();

  for (int k = tid; k < S_; k += 256) {
    const float* kr = K + ((size_t)(b * S_ + k)) * H_ + h * HD_;
    float s = 0.0f;
    #pragma unroll 8
    for (int d = 0; d < HD_; ++d) s += qv[d] * kr[d];
    s *= SCALE_ATT;
    if (src[b * S_ + k] == 0) s = -1e9f;   // PAD mask
    sc[k] = s;
  }
  __syncthreads();

  float lm = fmaxf(sc[tid], sc[tid + 256]);
  red[tid] = lm; __syncthreads();
  for (int st = 128; st > 0; st >>= 1) { if (tid < st) red[tid] = fmaxf(red[tid], red[tid + st]); __syncthreads(); }
  float mx = red[0]; __syncthreads();

  float e0 = expf(sc[tid] - mx), e1 = expf(sc[tid + 256] - mx);
  sc[tid] = e0; sc[tid + 256] = e1;
  red[tid] = e0 + e1; __syncthreads();
  for (int st = 128; st > 0; st >>= 1) { if (tid < st) red[tid] += red[tid + st]; __syncthreads(); }
  float inv = 1.0f / red[0];

  int d = tid & (HD_ - 1);
  int gi = tid >> 6;           // 4 k-groups
  float acc = 0.0f;
  for (int k = gi; k < S_; k += 4)
    acc += sc[k] * Vv[((size_t)(b * S_ + k)) * H_ + h * HD_ + d];
  part[gi][d] = acc; __syncthreads();
  if (gi == 0)
    O[((size_t)(b * S_ + qi)) * H_ + h * HD_ + d] =
        (part[0][d] + part[1][d] + part[2][d] + part[3][d]) * inv;
}

// =====================================================================
// Decoder cross-attention: 1 query per (b,h), over cached K/V [B,S,H].
// No mask (matches the reference). q: [B,H], O: [B,H].
// =====================================================================
__global__ __launch_bounds__(256)
void attn_cross_kernel(const float* __restrict__ q, const float* __restrict__ Kc,
                       const float* __restrict__ Vc, float* __restrict__ O)
{
  const int b = blockIdx.x / NH_;
  const int h = blockIdx.x % NH_;
  const int tid = threadIdx.x;
  __shared__ float qv[HD_];
  __shared__ float sc[S_];
  __shared__ float red[256];
  __shared__ float part[4][HD_];

  if (tid < HD_) qv[tid] = q[(size_t)b * H_ + h * HD_ + tid];
  __syncthreads();

  for (int k = tid; k < S_; k += 256) {
    const float* kr = Kc + ((size_t)(b * S_ + k)) * H_ + h * HD_;
    float s = 0.0f;
    #pragma unroll 8
    for (int d = 0; d < HD_; ++d) s += qv[d] * kr[d];
    sc[k] = s * SCALE_ATT;
  }
  __syncthreads();

  float lm = fmaxf(sc[tid], sc[tid + 256]);
  red[tid] = lm; __syncthreads();
  for (int st = 128; st > 0; st >>= 1) { if (tid < st) red[tid] = fmaxf(red[tid], red[tid + st]); __syncthreads(); }
  float mx = red[0]; __syncthreads();

  float e0 = expf(sc[tid] - mx), e1 = expf(sc[tid + 256] - mx);
  sc[tid] = e0; sc[tid + 256] = e1;
  red[tid] = e0 + e1; __syncthreads();
  for (int st = 128; st > 0; st >>= 1) { if (tid < st) red[tid] += red[tid + st]; __syncthreads(); }
  float inv = 1.0f / red[0];

  int d = tid & (HD_ - 1);
  int gi = tid >> 6;
  float acc = 0.0f;
  for (int k = gi; k < S_; k += 4)
    acc += sc[k] * Vc[((size_t)(b * S_ + k)) * H_ + h * HD_ + d];
  part[gi][d] = acc; __syncthreads();
  if (gi == 0)
    O[(size_t)b * H_ + h * HD_ + d] =
        (part[0][d] + part[1][d] + part[2][d] + part[3][d]) * inv;
}

// =====================================================================
// Row argmax over V_ logits, first-index tie-break (matches jnp.argmax).
// =====================================================================
__global__ __launch_bounds__(256)
void argmax_kernel(const float* __restrict__ logits, long long ld, int* __restrict__ tok)
{
  __shared__ float rv[256];
  __shared__ int   ri[256];
  int b = blockIdx.x, tid = threadIdx.x;
  const float* row = logits + (long long)b * ld;
  float bv = -3.4e38f; int bi = 0;
  for (int i = tid; i < V_; i += 256) {
    float x = row[i];
    if (x > bv) { bv = x; bi = i; }
  }
  rv[tid] = bv; ri[tid] = bi; __syncthreads();
  for (int st = 128; st > 0; st >>= 1) {
    if (tid < st) {
      if (rv[tid + st] > rv[tid] || (rv[tid + st] == rv[tid] && ri[tid + st] < ri[tid])) {
        rv[tid] = rv[tid + st]; ri[tid] = ri[tid + st];
      }
    }
    __syncthreads();
  }
  if (tid == 0) tok[b] = ri[0];
}

__global__ void zero_t0_kernel(float* __restrict__ out)
{
  int i = blockIdx.x * 256 + threadIdx.x;
  if (i < BB_ * V_) {
    int b = i / V_, n = i % V_;
    out[(size_t)b * MAXDEC_ * V_ + n] = 0.0f;
  }
}

__global__ void set_tok_kernel(int* __restrict__ tok)
{
  if (threadIdx.x < BB_) tok[threadIdx.x] = 1;   // SOS
}

// =====================================================================
// Host orchestration
// =====================================================================
extern "C" void kernel_launch(void* const* d_in, const int* in_sizes, int n_in,
                              void* d_out, int out_size, void* d_ws, size_t ws_size,
                              hipStream_t stream)
{
  (void)in_sizes; (void)n_in; (void)out_size; (void)ws_size;

  const int*   src         = (const int*)  d_in[0];
  const float* emb         = (const float*)d_in[1];
  const float* enc_attn_w  = (const float*)d_in[2];
  const float* enc_attn_b  = (const float*)d_in[3];
  const float* enc_ffn_w1  = (const float*)d_in[4];
  const float* enc_ffn_b1  = (const float*)d_in[5];
  const float* enc_ffn_w2  = (const float*)d_in[6];
  const float* enc_ffn_b2  = (const float*)d_in[7];
  const float* enc_ln      = (const float*)d_in[8];
  const float* dec_self_w  = (const float*)d_in[9];
  const float* dec_self_b  = (const float*)d_in[10];
  const float* dec_cross_w = (const float*)d_in[11];
  const float* dec_cross_b = (const float*)d_in[12];
  const float* dec_ffn_w1  = (const float*)d_in[13];
  const float* dec_ffn_b1  = (const float*)d_in[14];
  const float* dec_ffn_w2  = (const float*)d_in[15];
  const float* dec_ffn_b2  = (const float*)d_in[16];
  const float* dec_ln      = (const float*)d_in[17];
  const float* out_w       = (const float*)d_in[18];
  const float* out_b       = (const float*)d_in[19];
  float* out = (float*)d_out;

  // workspace carve-up (floats)
  float* ws = (float*)d_ws;
  size_t off = 0;
  auto alloc = [&](size_t n) { float* p = ws + off; off += n; return p; };
  float* x   = alloc((size_t)BS_ * H_);
  float* qb  = alloc((size_t)BS_ * H_);
  float* kb  = alloc((size_t)BS_ * H_);
  float* vb  = alloc((size_t)BS_ * H_);
  float* t1  = alloc((size_t)BS_ * H_);
  float* t2  = alloc((size_t)BS_ * H_);
  float* t0  = alloc((size_t)BS_ * FF_);
  float* kc  = alloc((size_t)LD_ * BS_ * H_);
  float* vc  = alloc((size_t)LD_ * BS_ * H_);
  float* y   = alloc(BB_ * H_);
  float* tv  = alloc(BB_ * H_);
  float* tsa = alloc(BB_ * H_);
  float* yq  = alloc(BB_ * H_);
  float* tco = alloc(BB_ * H_);
  float* to_ = alloc(BB_ * H_);
  float* h1  = alloc((size_t)BB_ * FF_);
  int*   tok = (int*)alloc(64);

  auto gemm = [&](const float* A, int lda, const float* Bw, int ldb,
                  const float* bias, float* Cp, long long ldc,
                  int M, int N, int K, int relu) {
    dim3 grid((N + BN_ - 1) / BN_, (M + BM_ - 1) / BM_);
    gemm_bf16_wmma_kernel<<<grid, 256, 0, stream>>>(A, lda, Bw, ldb, bias, Cp, ldc, M, N, K, relu);
  };

  zero_t0_kernel<<<(BB_ * V_ + 255) / 256, 256, 0, stream>>>(out);
  set_tok_kernel<<<1, 32, 0, stream>>>(tok);

  // ---------------- encoder ----------------
  embed_kernel<<<BS_, 128, 0, stream>>>(src, emb, x);
  for (int l = 0; l < LE_; ++l) {
    const float* Wq = enc_attn_w + ((size_t)l * 4 + 0) * H_ * H_;
    const float* Wk = enc_attn_w + ((size_t)l * 4 + 1) * H_ * H_;
    const float* Wv = enc_attn_w + ((size_t)l * 4 + 2) * H_ * H_;
    const float* Wo = enc_attn_w + ((size_t)l * 4 + 3) * H_ * H_;
    const float* bq = enc_attn_b + ((size_t)l * 4 + 0) * H_;
    const float* bk2 = enc_attn_b + ((size_t)l * 4 + 1) * H_;
    const float* bv2 = enc_attn_b + ((size_t)l * 4 + 2) * H_;
    const float* bo2 = enc_attn_b + ((size_t)l * 4 + 3) * H_;

    gemm(x, H_, Wq, H_, bq,  qb, H_, BS_, H_, H_, 0);
    gemm(x, H_, Wk, H_, bk2, kb, H_, BS_, H_, H_, 0);
    gemm(x, H_, Wv, H_, bv2, vb, H_, BS_, H_, H_, 0);
    attn_enc_kernel<<<dim3(BB_ * NH_, S_), 256, 0, stream>>>(qb, kb, vb, src, t1);
    gemm(t1, H_, Wo, H_, bo2, t2, H_, BS_, H_, H_, 0);
    add_ln_kernel<<<BS_, 128, 0, stream>>>(x, t2,
        enc_ln + (((size_t)l * 2 + 0) * 2 + 0) * H_,
        enc_ln + (((size_t)l * 2 + 0) * 2 + 1) * H_, x);

    gemm(x, H_, enc_ffn_w1 + (size_t)l * H_ * FF_, FF_, enc_ffn_b1 + (size_t)l * FF_,
         t0, FF_, BS_, FF_, H_, 1);
    gemm(t0, FF_, enc_ffn_w2 + (size_t)l * FF_ * H_, H_, enc_ffn_b2 + (size_t)l * H_,
         t2, H_, BS_, H_, FF_, 0);
    add_ln_kernel<<<BS_, 128, 0, stream>>>(x, t2,
        enc_ln + (((size_t)l * 2 + 1) * 2 + 0) * H_,
        enc_ln + (((size_t)l * 2 + 1) * 2 + 1) * H_, x);
  }

  // ---------------- cross-attn K/V cache (enc_out is fixed) ----------------
  for (int l = 0; l < LD_; ++l) {
    gemm(x, H_, dec_cross_w + ((size_t)l * 4 + 1) * H_ * H_, H_,
         dec_cross_b + ((size_t)l * 4 + 1) * H_, kc + (size_t)l * BS_ * H_, H_, BS_, H_, H_, 0);
    gemm(x, H_, dec_cross_w + ((size_t)l * 4 + 2) * H_ * H_, H_,
         dec_cross_b + ((size_t)l * 4 + 2) * H_, vc + (size_t)l * BS_ * H_, H_, BS_, H_, H_, 0);
  }

  // ---------------- greedy decode loop ----------------
  for (int t = 1; t < MAXDEC_; ++t) {
    embed_kernel<<<BB_, 128, 0, stream>>>(tok, emb, y);
    for (int l = 0; l < LD_; ++l) {
      // self-attention over a length-1 sequence == V then O projection
      gemm(y,  H_, dec_self_w + ((size_t)l * 4 + 2) * H_ * H_, H_,
           dec_self_b + ((size_t)l * 4 + 2) * H_, tv,  H_, BB_, H_, H_, 0);
      gemm(tv, H_, dec_self_w + ((size_t)l * 4 + 3) * H_ * H_, H_,
           dec_self_b + ((size_t)l * 4 + 3) * H_, tsa, H_, BB_, H_, H_, 0);
      add_ln_kernel<<<BB_, 128, 0, stream>>>(y, tsa,
          dec_ln + (((size_t)l * 3 + 0) * 2 + 0) * H_,
          dec_ln + (((size_t)l * 3 + 0) * 2 + 1) * H_, y);

      // cross-attention against cached K/V
      gemm(y, H_, dec_cross_w + ((size_t)l * 4 + 0) * H_ * H_, H_,
           dec_cross_b + ((size_t)l * 4 + 0) * H_, yq, H_, BB_, H_, H_, 0);
      attn_cross_kernel<<<BB_ * NH_, 256, 0, stream>>>(
          yq, kc + (size_t)l * BS_ * H_, vc + (size_t)l * BS_ * H_, tco);
      gemm(tco, H_, dec_cross_w + ((size_t)l * 4 + 3) * H_ * H_, H_,
           dec_cross_b + ((size_t)l * 4 + 3) * H_, to_, H_, BB_, H_, H_, 0);
      add_ln_kernel<<<BB_, 128, 0, stream>>>(y, to_,
          dec_ln + (((size_t)l * 3 + 1) * 2 + 0) * H_,
          dec_ln + (((size_t)l * 3 + 1) * 2 + 1) * H_, y);

      // FFN
      gemm(y, H_, dec_ffn_w1 + (size_t)l * H_ * FF_, FF_, dec_ffn_b1 + (size_t)l * FF_,
           h1, FF_, BB_, FF_, H_, 1);
      gemm(h1, FF_, dec_ffn_w2 + (size_t)l * FF_ * H_, H_, dec_ffn_b2 + (size_t)l * H_,
           tsa, H_, BB_, H_, FF_, 0);
      add_ln_kernel<<<BB_, 128, 0, stream>>>(y, tsa,
          dec_ln + (((size_t)l * 3 + 2) * 2 + 0) * H_,
          dec_ln + (((size_t)l * 3 + 2) * 2 + 1) * H_, y);
    }
    // logits straight into d_out[:, t, :]
    float* lt = out + (size_t)t * V_;
    gemm(y, H_, out_w, V_, out_b, lt, (long long)MAXDEC_ * V_, BB_, V_, H_, 0);
    argmax_kernel<<<BB_, 256, 0, stream>>>(lt, (long long)MAXDEC_ * V_, tok);
  }
}
